// MultiHeadedAttention_51994874085869
// MI455X (gfx1250) — matrix operand order repaired and checked by reading, not compiled, not run on hardware
//
#include <hip/hip_runtime.h>
#include <math.h>

// ---------------------------------------------------------------------------
// Equivariant multi-head attention, restructured for CDNA5 WMMA (gfx1250).
// Precompute: q = f@Wq^T ; Kf_f/Vf_f/Cf_f = readout-weights contracted with
// features (packed directly into WMMA-B f16 layout). Main passes run the
// radial MLPs as 16x16x32 f16 WMMA chains with f32 accumulation.
// ---------------------------------------------------------------------------

typedef _Float16 half_t;
typedef __attribute__((ext_vector_type(16))) _Float16 v16h;
typedef __attribute__((ext_vector_type(8)))  _Float16 v8h;
typedef __attribute__((ext_vector_type(8)))  float    v8f;

#define ZB     2      // batch
#define NN     256    // points
#define CHN    16     // channels
#define NHEADS 2
#define NBK    10     // key/conv radial basis
#define NBV3   3      // value radial basis
#define HDIM   100    // radial hidden width
#define NT     7      // N tiles  (7*16 = 112 >= 100)
#define KCH    4      // K chunks (4*32 = 128 >= 100)
#define LD     128    // LDS activation row stride (halves)

// ---------------------------------------------------------------------------
// WMMA helpers (wave32; A/B use contiguous k = 16*(lane/16) + half)
// ---------------------------------------------------------------------------

__device__ __forceinline__ v8f zero8() {
  v8f c = {0.f, 0.f, 0.f, 0.f, 0.f, 0.f, 0.f, 0.f};
  return c;
}

__device__ __forceinline__ v8f wmma_f16(v16h a, v16h b, v8f c) {
  // (neg_a, A, neg_b, B, c_mod, C, reuse_a, reuse_b)
  return __builtin_amdgcn_wmma_f32_16x16x32_f16(false, a, false, b,
                                                (short)0, c, false, false);
}

__device__ __forceinline__ void lds_fence() {
  asm volatile("s_wait_dscnt 0" ::: "memory");
}

// fast swish via hardware v_rcp_f32 (activation-accuracy is sufficient)
__device__ __forceinline__ float fast_swish(float x) {
  return x * __builtin_amdgcn_rcpf(1.0f + __expf(-x));
}

// A operand (16x32 f16) from a row-major [16][LD] f16 LDS tile.
// lane<16 : M=lane,   halves 0-15 = K 0..15
// lane>=16: M=lane-16,halves 0-15 = K 16..31
__device__ __forceinline__ v16h load_A(const half_t* act, int kbase, int lane) {
  const int row  = lane & 15;
  const int koff = (lane < 16) ? 0 : 16;
  const half_t* r = act + row * LD + kbase + koff;
  v8h lo = *(const v8h*)(r);
  v8h hi = *(const v8h*)(r + 8);
  v16h a;
#pragma unroll
  for (int j = 0; j < 8; ++j) { a[j] = lo[j]; a[8 + j] = hi[j]; }
  return a;
}

// B operand (32x16 f16), packed contiguously: [lane][16 halves]
// lane<16 : N=lane,    K = half
// lane>=16: N=lane-16, K = 16+half
__device__ __forceinline__ v16h loadB16(const half_t* p, int lane) {
  return *(const v16h*)(p + lane * 16);
}

// C tile store with swish + f16 convert into [16][LD] LDS tile.
// lane<16: N=lane rows 0..7 ; lane>=16: N=lane-16 rows 8..15
__device__ __forceinline__ void store_swish(half_t* buf, int nbase, v8f c, int lane) {
  const int col = lane & 15;
  const int r0  = (lane < 16) ? 0 : 8;
#pragma unroll
  for (int j = 0; j < 8; ++j) {
    buf[(r0 + j) * LD + nbase + col] = (half_t)fast_swish(c[j]);
  }
}

// Cosine basis -> f16 rows of the A-input tile (cols 0..31, zero padded).
__device__ __forceinline__ void write_basis(half_t* buf, float dist, int lane, int nb) {
  const int row = lane & 15;
  if (lane < 16) {
    const float step = 5.0f / (float)(nb - 1);
#pragma unroll
    for (int k = 0; k < 16; ++k) {
      float v = 0.f;
      if (k < nb) {
        float diff = (dist - (float)k * step) / step;
        if (fabsf(diff) < 1.f) v = __cosf(1.57079632679f * diff);
      }
      buf[row * LD + k] = (half_t)v;
    }
  } else {
#pragma unroll
    for (int k = 16; k < 32; ++k) buf[row * LD + k] = (half_t)0.f;
  }
}

__device__ __forceinline__ float pair_dist(const float* xyz, int z, int a, int b) {
  float dx = xyz[(z * NN + b) * 3 + 0] - xyz[(z * NN + a) * 3 + 0];
  float dy = xyz[(z * NN + b) * 3 + 1] - xyz[(z * NN + a) * 3 + 1];
  float dz = xyz[(z * NN + b) * 3 + 2] - xyz[(z * NN + a) * 3 + 2];
  return sqrtf(dx * dx + dy * dy + dz * dz + 1e-12f);
}

// ---------------------------------------------------------------------------
// Packing kernels
// ---------------------------------------------------------------------------

__device__ __forceinline__ half_t packW(const float* W, int Kd, int Nd,
                                        int nt, int kc, int ln, int hf) {
  int n  = nt * 16 + (ln & 15);
  int kl = (ln < 16) ? hf : 16 + hf;
  int k  = kc * 32 + kl;
  float v = (k < Kd && n < Nd) ? W[k * Nd + n] : 0.f;
  return (half_t)v;
}

#define W0P_CNT (NHEADS * NT * 512)
#define W1P_CNT (NHEADS * NT * KCH * 512)
#define V0P_CNT (NHEADS * NT * 512)
#define C0P_CNT (NT * 512)
#define C1P_CNT (NT * KCH * 512)

__global__ void __launch_bounds__(256)
pack_weights_kernel(const float* __restrict__ K0, const float* __restrict__ K1,
                    const float* __restrict__ V0, const float* __restrict__ C0,
                    const float* __restrict__ C1,
                    half_t* __restrict__ W0p, half_t* __restrict__ W1p,
                    half_t* __restrict__ V0p, half_t* __restrict__ C0p,
                    half_t* __restrict__ C1p) {
  int t = blockIdx.x * 256 + threadIdx.x;
  const int total = W0P_CNT + W1P_CNT + V0P_CNT + C0P_CNT + C1P_CNT;
  if (t >= total) return;
  if (t < W0P_CNT) {
    int hf = t & 15, ln = (t >> 4) & 31, nt = (t >> 9) % NT, h = t / (NT * 512);
    W0p[t] = packW(K0 + h * NBK * HDIM, NBK, HDIM, nt, 0, ln, hf);
    return;
  }
  t -= W0P_CNT;
  if (t < W1P_CNT) {
    int hf = t & 15, ln = (t >> 4) & 31, kc = (t >> 9) & 3;
    int nt = (t >> 11) % NT, h = t / (NT * KCH * 512);
    W1p[t] = packW(K1 + h * HDIM * HDIM, HDIM, HDIM, nt, kc, ln, hf);
    return;
  }
  t -= W1P_CNT;
  if (t < V0P_CNT) {
    int hf = t & 15, ln = (t >> 4) & 31, nt = (t >> 9) % NT, h = t / (NT * 512);
    V0p[t] = packW(V0 + h * NBV3 * HDIM, NBV3, HDIM, nt, 0, ln, hf);
    return;
  }
  t -= V0P_CNT;
  if (t < C0P_CNT) {
    int hf = t & 15, ln = (t >> 4) & 31, nt = t >> 9;
    C0p[t] = packW(C0, NBK, HDIM, nt, 0, ln, hf);
    return;
  }
  t -= C0P_CNT;
  {
    int hf = t & 15, ln = (t >> 4) & 31, kc = (t >> 9) & 3, nt = t >> 11;
    C1p[t] = packW(C1, HDIM, HDIM, nt, kc, ln, hf);
  }
}

// Kf_f[b,c,i] = sum_j Kf[h][c][i*16+j] * f[z,b,j]  (packed to WMMA-B layout)
__global__ void __launch_bounds__(256)
pack_feat_kernel(const float* __restrict__ features,
                 const float* __restrict__ Kf, const float* __restrict__ Vf,
                 half_t* __restrict__ KfP, half_t* __restrict__ VfP) {
  int t = blockIdx.x * 256 + threadIdx.x;
  if (t >= (1 << 21)) return;
  int hf = t & 15;
  int ln = (t >> 4) & 31;
  int kc = (t >> 9) & 3;
  int b  = (t >> 11) & 255;
  int h  = (t >> 19) & 1;
  int z  = (t >> 20) & 1;
  int i  = ln & 15;
  int kl = (ln < 16) ? hf : 16 + hf;
  int c  = kc * 32 + kl;
  float kv = 0.f, vv = 0.f;
  if (c < HDIM) {
    const float* f    = features + (z * NN + b) * CHN;
    const float* krow = Kf + ((size_t)(h * HDIM + c)) * (CHN * CHN) + i * CHN;
    const float* vrow = Vf + ((size_t)(h * HDIM + c)) * (CHN * CHN) + i * CHN;
#pragma unroll
    for (int j = 0; j < CHN; ++j) { kv += krow[j] * f[j]; vv += vrow[j] * f[j]; }
  }
  KfP[t] = (half_t)kv;
  VfP[t] = (half_t)vv;
}

// Cf_f[b,c,i] from post-attention features out_mid
__global__ void __launch_bounds__(256)
pack_out_kernel(const float* __restrict__ out_mid, const float* __restrict__ Cf,
                half_t* __restrict__ CfP) {
  int t = blockIdx.x * 256 + threadIdx.x;
  if (t >= (1 << 20)) return;
  int hf = t & 15;
  int ln = (t >> 4) & 31;
  int kc = (t >> 9) & 3;
  int b  = (t >> 11) & 255;
  int z  = (t >> 19) & 1;
  int i  = ln & 15;
  int kl = (ln < 16) ? hf : 16 + hf;
  int c  = kc * 32 + kl;
  float v = 0.f;
  if (c < HDIM) {
    const float* o    = out_mid + (z * NN + b) * CHN;
    const float* crow = Cf + ((size_t)c) * (CHN * CHN) + i * CHN;
#pragma unroll
    for (int j = 0; j < CHN; ++j) v += crow[j] * o[j];
  }
  CfP[t] = (half_t)v;
}

// q[z,h,a,o] = sum_i f[z,a,i] * Wq[h,o,i]
__global__ void __launch_bounds__(256)
qk_kernel(const float* __restrict__ features, const float* __restrict__ Wq,
          float* __restrict__ qbuf) {
  int t = blockIdx.x * 256 + threadIdx.x;
  if (t >= ZB * NHEADS * NN * CHN) return;
  int o = t & 15;
  int a = (t >> 4) & 255;
  int h = (t >> 12) & 1;
  int z = (t >> 13) & 1;
  float s = 0.f;
#pragma unroll
  for (int i = 0; i < CHN; ++i)
    s += features[(z * NN + a) * CHN + i] * Wq[(h * CHN + o) * CHN + i];
  qbuf[t] = s;
}

// ---------------------------------------------------------------------------
// Pass 1: scores[z,h,a,b] via radial2 MLP (WMMA) + key GEMM + q dot
// grid (bt=16, at=16, z=2), 128 threads = 4 waves, wave handles 4 b's
// ---------------------------------------------------------------------------
__global__ void __launch_bounds__(128)
pass1_scores_kernel(const float* __restrict__ xyz, const float* __restrict__ qbuf,
                    const half_t* __restrict__ W0p, const half_t* __restrict__ W1p,
                    const half_t* __restrict__ KfP, float* __restrict__ scores) {
  __shared__ __align__(16) half_t smem[4][2][16 * LD];
  {
    half_t* s = &smem[0][0][0];
    for (int idx = threadIdx.x; idx < 4 * 2 * 16 * LD; idx += 128)
      s[idx] = (half_t)0.f;
  }
  __syncthreads();

  const int lane = threadIdx.x & 31;
  const int wave = threadIdx.x >> 5;
  const int bt = blockIdx.x, at = blockIdx.y, z = blockIdx.z;
  const int aBase = at * 16;
  half_t* ping = smem[wave][0];
  half_t* pong = smem[wave][1];
  const int r0   = (lane < 16) ? 0 : 8;
  const int icol = lane & 15;

  for (int bi = wave; bi < 16; bi += 4) {
    const int b = bt * 16 + bi;
    float dist = 0.f;
    if (lane < 16) dist = pair_dist(xyz, z, aBase + lane, b);

    // prefetch both heads' packed Kf_f[b] (4KB each) under MLP compute
    {
      const half_t* kf0 = KfP + (((size_t)(z * NHEADS + 0) * NN + b) * KCH << 9);
      const half_t* kf1 = KfP + (((size_t)(z * NHEADS + 1) * NN + b) * KCH << 9);
      __builtin_prefetch(kf0 + lane * 64, 0, 1);
      __builtin_prefetch(kf1 + lane * 64, 0, 1);
    }

    for (int h = 0; h < NHEADS; ++h) {
      write_basis(ping, dist, lane, NBK);
      lds_fence();
      // layer 1: [16x10(pad32)] @ [10x100] -> swish ; A reused across N tiles
      {
        const v16h A = load_A(ping, 0, lane);
        for (int nt = 0; nt < NT; ++nt) {
          v8f c = zero8();
          c = wmma_f16(A, loadB16(W0p + ((h * NT + nt) << 9), lane), c);
          store_swish(pong, nt * 16, c, lane);
        }
      }
      lds_fence();
      // layer 2: [16x100(pad128)] @ [100x100] -> swish ; A chunks in regs
      {
        const v16h A0 = load_A(pong, 0, lane);
        const v16h A1 = load_A(pong, 32, lane);
        const v16h A2 = load_A(pong, 64, lane);
        const v16h A3 = load_A(pong, 96, lane);
        for (int nt = 0; nt < NT; ++nt) {
          const half_t* Wb = W1p + (((h * NT + nt) * KCH) << 9);
          v8f c = zero8();
          c = wmma_f16(A0, loadB16(Wb, lane), c);
          c = wmma_f16(A1, loadB16(Wb + 512, lane), c);
          c = wmma_f16(A2, loadB16(Wb + 1024, lane), c);
          c = wmma_f16(A3, loadB16(Wb + 1536, lane), c);
          store_swish(ping, nt * 16, c, lane);
        }
      }
      lds_fence();
      // key_t[a,i] = h2[a,:] @ Kf_f[b,:,i]
      const half_t* Kfb = KfP + (((size_t)(z * NHEADS + h) * NN + b) * KCH << 9);
      v8f kt = zero8();
      kt = wmma_f16(load_A(ping, 0, lane), loadB16(Kfb, lane), kt);
      kt = wmma_f16(load_A(ping, 32, lane), loadB16(Kfb + 512, lane), kt);
      kt = wmma_f16(load_A(ping, 64, lane), loadB16(Kfb + 1024, lane), kt);
      kt = wmma_f16(load_A(ping, 96, lane), loadB16(Kfb + 1536, lane), kt);
      // score[a] = key_t[a,:] . q[a,:] / CHN  (reduce over i = lane halves)
#pragma unroll
      for (int j = 0; j < 8; ++j) {
        int a = aBase + r0 + j;
        float s = kt[j] * qbuf[((size_t)(z * NHEADS + h) * NN + a) * CHN + icol];
        s += __shfl_xor(s, 1, 32);
        s += __shfl_xor(s, 2, 32);
        s += __shfl_xor(s, 4, 32);
        s += __shfl_xor(s, 8, 32);
        if (icol == 0)
          scores[((size_t)(z * NHEADS + h) * NN + a) * NN + b] = s * (1.0f / CHN);
      }
    }
  }
}

// ---------------------------------------------------------------------------
// Softmax over b: one wave per row of 256
// ---------------------------------------------------------------------------
__global__ void __launch_bounds__(256)
softmax_kernel(float* __restrict__ scores) {
  const int lane = threadIdx.x & 31;
  const int wave = threadIdx.x >> 5;
  const int row = blockIdx.x * 8 + wave;
  if (row >= ZB * NHEADS * NN) return;
  float* sr = scores + (size_t)row * NN;
  float v[8];
  float m = -3.4e38f;
#pragma unroll
  for (int k = 0; k < 8; ++k) { v[k] = sr[lane + 32 * k]; m = fmaxf(m, v[k]); }
#pragma unroll
  for (int off = 16; off >= 1; off >>= 1) m = fmaxf(m, __shfl_xor(m, off, 32));
  float s = 0.f;
#pragma unroll
  for (int k = 0; k < 8; ++k) { v[k] = __expf(v[k] - m); s += v[k]; }
#pragma unroll
  for (int off = 16; off >= 1; off >>= 1) s += __shfl_xor(s, off, 32);
  const float inv = __builtin_amdgcn_rcpf(s);
#pragma unroll
  for (int k = 0; k < 8; ++k) sr[lane + 32 * k] = v[k] * inv;
}

// ---------------------------------------------------------------------------
// Pass 2: out_mid[a] = sum_h sum_b p[a,b] * (hv[a,b,:] @ Vf_f[b,:,:])
// grid (bc=64, at=16, z=2), 1 wave per block, 4 b's per wave.
// Deterministic: partials per b-chunk, fixed-order reduction afterwards.
// ---------------------------------------------------------------------------
__global__ void __launch_bounds__(32)
pass2_value_kernel(const float* __restrict__ xyz, const float* __restrict__ probs,
                   const half_t* __restrict__ V0p, const half_t* __restrict__ VfP,
                   float* __restrict__ part2) {
  __shared__ __align__(16) half_t ping[16 * LD];
  __shared__ __align__(16) half_t pong[16 * LD];
  const int lane = threadIdx.x;
  for (int idx = lane; idx < 16 * LD; idx += 32) {
    ping[idx] = (half_t)0.f;
    pong[idx] = (half_t)0.f;
  }
  lds_fence();
  const int bc = blockIdx.x, at = blockIdx.y, z = blockIdx.z;
  const int aBase = at * 16;
  const int r0   = (lane < 16) ? 0 : 8;
  const int icol = lane & 15;
  v8f acc = zero8();

  for (int k4 = 0; k4 < 4; ++k4) {
    const int b = bc * 4 + k4;
    float dist = 0.f;
    if (lane < 16) dist = pair_dist(xyz, z, aBase + lane, b);
    for (int h = 0; h < NHEADS; ++h) {
      const half_t* Vfb = VfP + (((size_t)(z * NHEADS + h) * NN + b) * KCH << 9);
      __builtin_prefetch(Vfb + lane * 64, 0, 1);
      write_basis(ping, dist, lane, NBV3);
      lds_fence();
      {
        const v16h A = load_A(ping, 0, lane);
        for (int nt = 0; nt < NT; ++nt) {
          v8f c = zero8();
          c = wmma_f16(A, loadB16(V0p + ((h * NT + nt) << 9), lane), c);
          store_swish(pong, nt * 16, c, lane);
        }
      }
      lds_fence();
      v8f vt = zero8();
      vt = wmma_f16(load_A(pong, 0, lane), loadB16(Vfb, lane), vt);
      vt = wmma_f16(load_A(pong, 32, lane), loadB16(Vfb + 512, lane), vt);
      vt = wmma_f16(load_A(pong, 64, lane), loadB16(Vfb + 1024, lane), vt);
      vt = wmma_f16(load_A(pong, 96, lane), loadB16(Vfb + 1536, lane), vt);
#pragma unroll
      for (int j = 0; j < 8; ++j) {
        int a = aBase + r0 + j;
        float pv = probs[((size_t)(z * NHEADS + h) * NN + a) * NN + b];
        acc[j] += vt[j] * pv;
      }
    }
  }
#pragma unroll
  for (int j = 0; j < 8; ++j)
    part2[(((size_t)z * 64 + bc) * NN + (aBase + r0 + j)) * CHN + icol] = acc[j];
}

// ---------------------------------------------------------------------------
// Pass 3: result[a] = sum_b hc2[a,b,:] @ Cf_f[b,:,:]   (final conv)
// ---------------------------------------------------------------------------
__global__ void __launch_bounds__(32)
pass3_conv_kernel(const float* __restrict__ xyz,
                  const half_t* __restrict__ C0p, const half_t* __restrict__ C1p,
                  const half_t* __restrict__ CfP, float* __restrict__ part3) {
  __shared__ __align__(16) half_t ping[16 * LD];
  __shared__ __align__(16) half_t pong[16 * LD];
  const int lane = threadIdx.x;
  for (int idx = lane; idx < 16 * LD; idx += 32) {
    ping[idx] = (half_t)0.f;
    pong[idx] = (half_t)0.f;
  }
  lds_fence();
  const int bc = blockIdx.x, at = blockIdx.y, z = blockIdx.z;
  const int aBase = at * 16;
  const int r0   = (lane < 16) ? 0 : 8;
  const int icol = lane & 15;
  v8f acc = zero8();

  for (int k4 = 0; k4 < 4; ++k4) {
    const int b = bc * 4 + k4;
    float dist = 0.f;
    if (lane < 16) dist = pair_dist(xyz, z, aBase + lane, b);
    const half_t* Cfb = CfP + (((size_t)z * NN + b) * KCH << 9);
    __builtin_prefetch(Cfb + lane * 64, 0, 1);

    write_basis(ping, dist, lane, NBK);
    lds_fence();
    {
      const v16h A = load_A(ping, 0, lane);
      for (int nt = 0; nt < NT; ++nt) {
        v8f c = zero8();
        c = wmma_f16(A, loadB16(C0p + (nt << 9), lane), c);
        store_swish(pong, nt * 16, c, lane);
      }
    }
    lds_fence();
    {
      const v16h A0 = load_A(pong, 0, lane);
      const v16h A1 = load_A(pong, 32, lane);
      const v16h A2 = load_A(pong, 64, lane);
      const v16h A3 = load_A(pong, 96, lane);
      for (int nt = 0; nt < NT; ++nt) {
        const half_t* Wb = C1p + ((nt * KCH) << 9);
        v8f c = zero8();
        c = wmma_f16(A0, loadB16(Wb, lane), c);
        c = wmma_f16(A1, loadB16(Wb + 512, lane), c);
        c = wmma_f16(A2, loadB16(Wb + 1024, lane), c);
        c = wmma_f16(A3, loadB16(Wb + 1536, lane), c);
        store_swish(ping, nt * 16, c, lane);
      }
    }
    lds_fence();
    v8f ct = zero8();
    ct = wmma_f16(load_A(ping, 0, lane), loadB16(Cfb, lane), ct);
    ct = wmma_f16(load_A(ping, 32, lane), loadB16(Cfb + 512, lane), ct);
    ct = wmma_f16(load_A(ping, 64, lane), loadB16(Cfb + 1024, lane), ct);
    ct = wmma_f16(load_A(ping, 96, lane), loadB16(Cfb + 1536, lane), ct);
    acc = acc + ct;
  }
#pragma unroll
  for (int j = 0; j < 8; ++j)
    part3[(((size_t)z * 64 + bc) * NN + (aBase + r0 + j)) * CHN + icol] = acc[j];
}

// Fixed-order reduction over the 64 b-chunk slices (deterministic).
__global__ void __launch_bounds__(256)
reduce_slices_kernel(const float* __restrict__ part, float* __restrict__ dst) {
  int t = blockIdx.x * 256 + threadIdx.x;
  if (t >= ZB * NN * CHN) return;
  int z = t / (NN * CHN);
  int rest = t - z * (NN * CHN);
  float s = 0.f;
  for (int c = 0; c < 64; ++c)
    s += part[((size_t)(z * 64 + c)) * (NN * CHN) + rest];
  dst[t] = s;
}

// ---------------------------------------------------------------------------
// Host launcher
// ---------------------------------------------------------------------------
extern "C" void kernel_launch(void* const* d_in, const int* in_sizes, int n_in,
                              void* d_out, int out_size, void* d_ws, size_t ws_size,
                              hipStream_t stream) {
  (void)in_sizes; (void)n_in; (void)ws_size;
  const float* features = (const float*)d_in[0];
  const float* xyz      = (const float*)d_in[1];
  const float* Wq       = (const float*)d_in[2];
  const float* K0       = (const float*)d_in[3];
  const float* K1       = (const float*)d_in[4];
  const float* Kf       = (const float*)d_in[5];
  const float* V0       = (const float*)d_in[6];
  const float* Vf       = (const float*)d_in[7];
  const float* C0       = (const float*)d_in[8];
  const float* C1       = (const float*)d_in[9];
  const float* Cf       = (const float*)d_in[10];
  float* outF = (float*)d_out;
  (void)out_size;

  char* ws = (char*)d_ws;
  size_t off = 0;
  auto take = [&](size_t bytes) -> size_t {
    size_t o = off;
    off += (bytes + 255) & ~(size_t)255;
    return o;
  };
  float*  qbuf   = (float*)(ws + take((size_t)ZB * NHEADS * NN * CHN * 4));
  float*  scores = (float*)(ws + take((size_t)ZB * NHEADS * NN * NN * 4));
  float*  outmid = (float*)(ws + take((size_t)ZB * NN * CHN * 4));
  float*  part2  = (float*)(ws + take((size_t)ZB * 64 * NN * CHN * 4));
  float*  part3  = (float*)(ws + take((size_t)ZB * 64 * NN * CHN * 4));
  half_t* W0p    = (half_t*)(ws + take((size_t)W0P_CNT * 2));
  half_t* W1p    = (half_t*)(ws + take((size_t)W1P_CNT * 2));
  half_t* V0p    = (half_t*)(ws + take((size_t)V0P_CNT * 2));
  half_t* C0p    = (half_t*)(ws + take((size_t)C0P_CNT * 2));
  half_t* C1p    = (half_t*)(ws + take((size_t)C1P_CNT * 2));
  half_t* KfP    = (half_t*)(ws + take((size_t)(1 << 21) * 2));
  half_t* VfP    = (half_t*)(ws + take((size_t)(1 << 21) * 2));
  half_t* CfP    = (half_t*)(ws + take((size_t)(1 << 20) * 2));

  const int packw_total = W0P_CNT + W1P_CNT + V0P_CNT + C0P_CNT + C1P_CNT;

  pack_weights_kernel<<<(packw_total + 255) / 256, 256, 0, stream>>>(
      K0, K1, V0, C0, C1, W0p, W1p, V0p, C0p, C1p);
  pack_feat_kernel<<<(1 << 21) / 256, 256, 0, stream>>>(features, Kf, Vf, KfP, VfP);
  qk_kernel<<<(ZB * NHEADS * NN * CHN + 255) / 256, 256, 0, stream>>>(features, Wq, qbuf);

  pass1_scores_kernel<<<dim3(16, 16, ZB), 128, 0, stream>>>(xyz, qbuf, W0p, W1p, KfP, scores);
  softmax_kernel<<<(ZB * NHEADS * NN) / 8, 256, 0, stream>>>(scores);
  pass2_value_kernel<<<dim3(64, 16, ZB), 32, 0, stream>>>(xyz, scores, V0p, VfP, part2);
  reduce_slices_kernel<<<(ZB * NN * CHN + 255) / 256, 256, 0, stream>>>(part2, outmid);
  pack_out_kernel<<<(1 << 20) / 256, 256, 0, stream>>>(outmid, Cf, CfP);
  pass3_conv_kernel<<<dim3(64, 16, ZB), 32, 0, stream>>>(xyz, C0p, C1p, CfP, part3);
  reduce_slices_kernel<<<(ZB * NN * CHN + 255) / 256, 256, 0, stream>>>(part3, outF);
}
